// TFAHead_16484084483645
// MI455X (gfx1250) — matrix operand order repaired and checked by reading, not compiled, not run on hardware
//
#include <hip/hip_runtime.h>

typedef _Float16 h16;
typedef __attribute__((ext_vector_type(16))) _Float16 v16h;
typedef __attribute__((ext_vector_type(8)))  _Float16 v8h;
typedef __attribute__((ext_vector_type(4)))  _Float16 v4h;
typedef __attribute__((ext_vector_type(8)))  float    v8f;

#define NEG_BIG (-3.402823466e+38f)

// ---------------------------------------------------------------------------
// WMMA fragment helpers (CDNA5 v_wmma_f32_16x16x32_f16, wave32)
// A layout: lane 0-15 -> M=lane, K chunks {0..7, 16..23}; lane 16-31 -> M=lane-16,
//           K chunks {8..15, 24..31}.  Stored row-major [M][K] (K mult of 8).
// B layout: lane 0-15 -> N=lane, K chunks {0..7, 16..23}; lane 16-31 -> N=lane-16, K chunks {8..15, 24..31}.
//           Stored transposed [N][K] (K contiguous).
// C/D:      VGPR v -> M = v + 8*(lane>=16), N = lane&15.
// ---------------------------------------------------------------------------
__device__ __forceinline__ v16h load_afrag(const h16* base, int stride, int m,
                                           int k0, int lane) {
  const int boff = (lane < 16) ? 0 : 8;
  const h16* r = base + m * stride + k0 + boff;
  v8h lo = *(const v8h*)(r);
  v8h hi = *(const v8h*)(r + 16);
  v16h a;
#pragma unroll
  for (int i = 0; i < 8; ++i) { a[i] = lo[i]; a[i + 8] = hi[i]; }
  return a;
}

__device__ __forceinline__ v16h load_bfrag(const h16* baseT, int stride, int n,
                                           int k0, int lane) {
  const int koff = (lane < 16) ? 0 : 8;
  const h16* r = baseT + n * stride + k0 + koff;
  v8h lo = *(const v8h*)(r);
  v8h hi = *(const v8h*)(r + 16);
  v16h b;
#pragma unroll
  for (int i = 0; i < 8; ++i) { b[i] = lo[i]; b[i + 8] = hi[i]; }
  return b;
}

#define WMMA(a, b, c) \
  __builtin_amdgcn_wmma_f32_16x16x32_f16(false, (a), false, (b), (short)0, (c), false, false)

__device__ __forceinline__ float sigmoidf_(float x) { return 1.0f / (1.0f + __expf(-x)); }

__device__ __forceinline__ float block_reduce_sum(float x, float* red) {
#pragma unroll
  for (int off = 16; off >= 1; off >>= 1) x += __shfl_xor(x, off);
  const int tid = threadIdx.x;
  if ((tid & 31) == 0) red[tid >> 5] = x;
  __syncthreads();
  if (tid == 0) {
    float s = 0.f;
#pragma unroll
    for (int i = 0; i < 8; ++i) s += red[i];
    red[0] = s;
  }
  __syncthreads();
  float r = red[0];
  __syncthreads();
  return r;
}

// ---------------------------------------------------------------------------
// Stage 1: horizontal mean+max pool over W per H strip.
// x [B,C,T,H,W] -> z [P,B,C,T]  (P == H, strip height 1)
// One block per (b,c): stage the [T,H,W] slice in LDS with float4 loads
// (slice is contiguous & 16B aligned: 5280 floats), then reduce.
// ---------------------------------------------------------------------------
__global__ __launch_bounds__(256) void hpool_kernel(const float* __restrict__ x,
                                                    float* __restrict__ zg) {
  __shared__ __attribute__((aligned(16))) float xs[30 * 16 * 11]; // 5280 floats
  const int blk = blockIdx.x;
  const int b = blk >> 8, c = blk & 255;
  const float4* xp4 = (const float4*)(x + (size_t)(b * 256 + c) * (30 * 16 * 11));
  float4* xs4 = (float4*)xs;
  for (int i = threadIdx.x; i < 1320; i += 256) xs4[i] = xp4[i];
  __syncthreads();
  for (int j = threadIdx.x; j < 480; j += 256) {
    const int p = j / 30, t = j - p * 30;
    const float* r = &xs[(t * 16 + p) * 11];
    float s = 0.f, mx = NEG_BIG;
#pragma unroll
    for (int w = 0; w < 11; ++w) { float v = r[w]; s += v; mx = fmaxf(mx, v); }
    zg[((p * 64 + b) * 256 + c) * 30 + t] = s * (1.0f / 11.0f) + mx;
  }
}

// ---------------------------------------------------------------------------
// Utility: f32 -> f16 convert, 4 elements per thread (n multiple of 4)
// ---------------------------------------------------------------------------
__global__ __launch_bounds__(256) void cvt_f16_kernel(const float* __restrict__ s,
                                                      h16* __restrict__ d, int n4) {
  int i = blockIdx.x * 256 + threadIdx.x;
  if (i < n4) {
    float4 v = ((const float4*)s)[i];
    v4h h;
    h[0] = (h16)v.x; h[1] = (h16)v.y; h[2] = (h16)v.z; h[3] = (h16)v.w;
    *(v4h*)(d + 4 * i) = h;
  }
}

// FForward [p][c][o] f32 -> transposed f16 [p][o][c]
__global__ __launch_bounds__(256) void ff_transpose_kernel(const float* __restrict__ ff,
                                                           h16* __restrict__ fft) {
  int i = blockIdx.x * 256 + threadIdx.x; // 16*256*256
  int p = i >> 16, r = i & 65535, o = r >> 8, c = r & 255;
  fft[(p * 256 + o) * 256 + c] = (h16)ff[(p * 256 + c) * 256 + o];
}

// ---------------------------------------------------------------------------
// Stage 2-4: per-(p,b) tile. Gated convs as WMMA GEMMs + T-pools + max over T.
// Grid: 1024 blocks (p*64+b), 256 threads (8 waves).
// ---------------------------------------------------------------------------
__global__ __launch_bounds__(256) void gate_feat_kernel(
    const float* __restrict__ zg,
    const h16* __restrict__ w31a, const float* __restrict__ b31a,
    const h16* __restrict__ w31b, const float* __restrict__ b31b,
    const h16* __restrict__ w33a, const float* __restrict__ b33a,
    const h16* __restrict__ w33b, const float* __restrict__ b33b,
    h16* __restrict__ featg) {
  __shared__ __attribute__((aligned(16))) h16 zs[256 * 32];   // 16 KB  z tile (T padded to 32)
  __shared__ __attribute__((aligned(16))) h16 Bt1[32 * 384];  // 24 KB  im2col(z)^T, half-K
  __shared__ __attribute__((aligned(16))) h16 Bt31[32 * 64];  //  4 KB  h1_31^T
  __shared__ __attribute__((aligned(16))) h16 Bt33[32 * 192]; // 12 KB  im2col(h1_33)^T
  const int p = blockIdx.x >> 6, b = blockIdx.x & 63;
  const int tid = threadIdx.x, lane = tid & 31, wv = tid >> 5;

  // load z tile (contiguous 7680 floats, 16B aligned) via float4, pad T
  const float4* zpb4 =
      (const float4*)(zg + (size_t)((p * 64 + b) * 256) * 30);
  for (int i = tid; i < 1920; i += 256) {
    float4 v = zpb4[i];
    const int j = i * 4;
#pragma unroll
    for (int k = 0; k < 4; ++k) {
      int jj = j + k;
      int c = jj / 30, t = jj - c * 30;
      float vv = (k == 0) ? v.x : (k == 1) ? v.y : (k == 2) ? v.z : v.w;
      zs[c * 32 + t] = (h16)vv;
    }
  }
  { int c = tid; zs[c * 32 + 30] = (h16)0.f; zs[c * 32 + 31] = (h16)0.f; }
  for (int i = tid; i < 32 * 192; i += 256) Bt33[i] = (h16)0.f;
  __syncthreads();

  // ---- conv1 (both gates share B = im2col(z)); K=768 split into two halves
  v8f accA = {}; // this wave's gate31 tile
  v8f accB = {}; // this wave's gate33 tile
  const h16* wA31 = w31a + p * (64 * 768);
  const h16* wA33 = w33a + p * (64 * 768);
  for (int half = 0; half < 2; ++half) {
    for (int i = tid; i < 32 * 384; i += 256) {
      int t = i / 384, kk = i - t * 384, k = half * 384 + kk;
      int c = k / 3, dk = k - c * 3, tt = t + dk - 1;
      float v = (tt >= 0 && tt < 30) ? (float)zs[c * 32 + tt] : 0.f;
      Bt1[i] = (h16)v;
    }
    __syncthreads();
#pragma unroll
    for (int r = 0; r < 2; ++r) {
      const int q = wv + 8 * r;            // 0..7 gate31, 8..15 gate33
      const int qq = q & 7, mt = qq >> 1, nt = qq & 1;
      const h16* wA = (q < 8) ? wA31 : wA33;
      const int m = mt * 16 + (lane & 15);
      const int n = nt * 16 + (lane & 15);
      v8f acc = (r == 0) ? accA : accB;
      for (int ks = 0; ks < 12; ++ks) {
        v16h a = load_afrag(wA, 768, m, half * 384 + ks * 32, lane);
        v16h bb = load_bfrag(Bt1, 384, n, ks * 32, lane);
        acc = WMMA(a, bb, acc);
      }
      if (r == 0) accA = acc; else accB = acc;
    }
    __syncthreads();
  }

  // bias + leaky_relu, scatter h1 into conv2 B matrices (f16)
#pragma unroll
  for (int r = 0; r < 2; ++r) {
    const int q = wv + 8 * r;
    const int qq = q & 7, mt = qq >> 1, nt = qq & 1;
    const float* bias = ((q < 8) ? b31a : b33a) + p * 64;
    const int n = nt * 16 + (lane & 15);
    const int mb = mt * 16 + ((lane < 16) ? 0 : 8);
    v8f acc = (r == 0) ? accA : accB;
#pragma unroll
    for (int v = 0; v < 8; ++v) {
      const int m = mb + v;
      float x = acc[v] + bias[m];
      x = (x > 0.f) ? x : 0.01f * x;
      h16 hx = (h16)x;
      if (q < 8) {
        Bt31[n * 64 + m] = hx;
      } else {
#pragma unroll
        for (int dk = 0; dk < 3; ++dk) {
          int tp = n + 1 - dk;
          if (tp >= 0 && tp < 32) Bt33[tp * 192 + m * 3 + dk] = hx;
        }
      }
    }
  }
  __syncthreads();

  // ---- conv2 (both gates) + sigmoid + T-pools + combine + max over T
  const h16* wB31 = w31b + p * (256 * 64);
  const h16* wB33 = w33b + p * (256 * 192);
  const float* bb31 = b31b + p * 256;
  const float* bb33 = b33b + p * 256;
#pragma unroll
  for (int r = 0; r < 2; ++r) {
    const int mt = wv + 8 * r; // 16 M-tiles of C=256, exclusive per wave
    float fm[8];
#pragma unroll
    for (int v = 0; v < 8; ++v) fm[v] = NEG_BIG;
    for (int nt = 0; nt < 2; ++nt) {
      const int mrow = mt * 16 + (lane & 15);
      const int n = nt * 16 + (lane & 15);
      v8f a31 = {}, a33 = {};
      for (int ks = 0; ks < 2; ++ks) {
        v16h a = load_afrag(wB31, 64, mrow, ks * 32, lane);
        v16h bv = load_bfrag(Bt31, 64, n, ks * 32, lane);
        a31 = WMMA(a, bv, a31);
      }
      for (int ks = 0; ks < 6; ++ks) {
        v16h a = load_afrag(wB33, 192, mrow, ks * 32, lane);
        v16h bv = load_bfrag(Bt33, 192, n, ks * 32, lane);
        a33 = WMMA(a, bv, a33);
      }
      const int t = n;
      if (t < 30) {
        const int mb = mt * 16 + ((lane < 16) ? 0 : 8);
#pragma unroll
        for (int v = 0; v < 8; ++v) {
          const int c = mb + v;
          const float s31 = sigmoidf_(a31[v] + bb31[c]);
          const float s33 = sigmoidf_(a33[v] + bb33[c]);
          const h16* zr = &zs[c * 32];
          float sum3 = 0.f, mx3 = NEG_BIG, sum5 = 0.f, mx5 = NEG_BIG;
#pragma unroll
          for (int d = -2; d <= 2; ++d) {
            int u = t + d;
            if (u >= 0 && u < 30) {
              float zv = (float)zr[u];
              sum5 += zv; mx5 = fmaxf(mx5, zv);
              if (d >= -1 && d <= 1) { sum3 += zv; mx3 = fmaxf(mx3, zv); }
            }
          }
          const float a3v = sum3 * (1.f / 3.f) + mx3;
          const float a5v = sum5 * 0.2f + mx5;
          fm[v] = fmaxf(fm[v], a3v * s31 + a5v * s33);
        }
      }
    }
    // max across the 16 lanes of each half-wave (same row, different t)
#pragma unroll
    for (int v = 0; v < 8; ++v) {
      float x = fm[v];
      x = fmaxf(x, __shfl_xor(x, 1));
      x = fmaxf(x, __shfl_xor(x, 2));
      x = fmaxf(x, __shfl_xor(x, 4));
      x = fmaxf(x, __shfl_xor(x, 8));
      fm[v] = x;
    }
    if ((lane & 15) == 0) {
      const int mb = mt * 16 + ((lane < 16) ? 0 : 8);
#pragma unroll
      for (int v = 0; v < 8; ++v)
        featg[(size_t)((p * 64 + b) * 256) + mb + v] = (h16)fm[v];
    }
  }
}

// ---------------------------------------------------------------------------
// Stage 5: outputs[p] = feat[p] (64x256) x FForward[p] (256x256); write [b,p,o]
// Grid 64: p = blk>>2, chunk = blk&3; each wave does 2 of 64 tiles.
// ---------------------------------------------------------------------------
__global__ __launch_bounds__(256) void outputs_gemm_kernel(const h16* __restrict__ feat16,
                                                           const h16* __restrict__ ff16t,
                                                           float* __restrict__ outg) {
  const int p = blockIdx.x >> 2, chunk = blockIdx.x & 3;
  const int lane = threadIdx.x & 31, wv = threadIdx.x >> 5;
  const h16* A = feat16 + p * (64 * 256);
  const h16* Bt = ff16t + p * (256 * 256);
#pragma unroll
  for (int s = 0; s < 2; ++s) {
    const int q = chunk * 16 + wv * 2 + s; // 0..63
    const int mt = q & 3, nt = q >> 2;
    const int m = mt * 16 + (lane & 15);
    const int n = nt * 16 + (lane & 15);
    v8f acc = {};
    for (int ks = 0; ks < 8; ++ks) {
      v16h a = load_afrag(A, 256, m, ks * 32, lane);
      v16h bv = load_bfrag(Bt, 256, n, ks * 32, lane);
      acc = WMMA(a, bv, acc);
    }
    const int mb = mt * 16 + ((lane < 16) ? 0 : 8);
#pragma unroll
    for (int v = 0; v < 8; ++v)
      outg[(size_t)((mb + v) * 16 + p) * 256 + n] = acc[v];
  }
}

// ---------------------------------------------------------------------------
// Stage 6: per-part BatchNorm (batch stats over b) + L2 row norm; -> bnn f16
// Grid 16 (one block per part), 256 threads (one per o).
// ---------------------------------------------------------------------------
__global__ __launch_bounds__(256) void bn_norm_kernel(const float* __restrict__ outg,
                                                      const float* __restrict__ gamma,
                                                      const float* __restrict__ beta,
                                                      h16* __restrict__ bnn16) {
  __shared__ float As[256], Bs[256];
  __shared__ float red[8];
  const int p = blockIdx.x, o = threadIdx.x;
  float s1 = 0.f, s2 = 0.f;
  for (int bb = 0; bb < 64; ++bb) {
    float v = outg[(size_t)(bb * 16 + p) * 256 + o];
    s1 += v; s2 += v * v;
  }
  const float mu = s1 * (1.f / 64.f);
  const float var = s2 * (1.f / 64.f) - mu * mu;
  const float inv = rsqrtf(var + 1e-5f);
  const float g = gamma[p * 256 + o], be = beta[p * 256 + o];
  As[o] = inv * g;
  Bs[o] = be - mu * inv * g;
  __syncthreads();
  for (int bb = 0; bb < 64; ++bb) {
    float v = outg[(size_t)(bb * 16 + p) * 256 + o];
    float bn = v * As[o] + Bs[o];
    float ss = block_reduce_sum(bn * bn, red);
    float sc = 1.f / fmaxf(sqrtf(ss), 1e-12f);
    bnn16[(size_t)((p * 64 + bb) * 256) + o] = (h16)(bn * sc);
  }
}

// ---------------------------------------------------------------------------
// Stage 7: fc column L2-norm -> transposed f16 [p][n(1008 pad)][o]
// Grid 16*1008, 256 threads (one per o).
// ---------------------------------------------------------------------------
__global__ __launch_bounds__(256) void fc_norm_kernel(const float* __restrict__ fc,
                                                      h16* __restrict__ fcT) {
  __shared__ float red[8];
  const int blk = blockIdx.x;
  const int p = blk / 1008, n = blk - p * 1008;
  const int o = threadIdx.x;
  float v = (n < 1000) ? fc[(size_t)(p * 256 + o) * 1000 + n] : 0.f;
  float ss = block_reduce_sum(v * v, red);
  float sc = 1.f / fmaxf(sqrtf(ss), 1e-12f);
  fcT[(size_t)(p * 1008 + n) * 256 + o] = (h16)(v * sc);
}

// ---------------------------------------------------------------------------
// Stage 8: logits[p] = bnn[p] (64x256) x fcn[p] (256x1000); write [b,p,n]
// Grid 512: p = blk>>5, nc = blk&31; wave -> (mt = wv&3, nt = nc*2 + wv>>2)
// ---------------------------------------------------------------------------
__global__ __launch_bounds__(256) void logits_gemm_kernel(const h16* __restrict__ bnn16,
                                                          const h16* __restrict__ fcT,
                                                          float* __restrict__ logits) {
  const int p = blockIdx.x >> 5, nc = blockIdx.x & 31;
  const int lane = threadIdx.x & 31, wv = threadIdx.x >> 5;
  const int mt = wv & 3, nt = nc * 2 + (wv >> 2);
  if (nt >= 63) return;
  const h16* A = bnn16 + p * (64 * 256);
  const h16* Bt = fcT + (size_t)p * (1008 * 256);
  const int m = mt * 16 + (lane & 15);
  const int n = nt * 16 + (lane & 15);
  v8f acc = {};
  for (int ks = 0; ks < 8; ++ks) {
    v16h a = load_afrag(A, 256, m, ks * 32, lane);
    v16h bv = load_bfrag(Bt, 256, n, ks * 32, lane);
    acc = WMMA(a, bv, acc);
  }
  const int mb = mt * 16 + ((lane < 16) ? 0 : 8);
#pragma unroll
  for (int v = 0; v < 8; ++v) {
    if (n < 1000)
      logits[(size_t)((mb + v) * 16 + p) * 1000 + n] = acc[v];
  }
}

// ---------------------------------------------------------------------------
// Host launcher
// ---------------------------------------------------------------------------
extern "C" void kernel_launch(void* const* d_in, const int* in_sizes, int n_in,
                              void* d_out, int out_size, void* d_ws, size_t ws_size,
                              hipStream_t stream) {
  (void)in_sizes; (void)n_in; (void)out_size; (void)ws_size;
  const float* x     = (const float*)d_in[0];
  const float* w31a  = (const float*)d_in[1];
  const float* b31a  = (const float*)d_in[2];
  const float* w31b  = (const float*)d_in[3];
  const float* b31b  = (const float*)d_in[4];
  const float* w33a  = (const float*)d_in[5];
  const float* b33a  = (const float*)d_in[6];
  const float* w33b  = (const float*)d_in[7];
  const float* b33b  = (const float*)d_in[8];
  const float* FF    = (const float*)d_in[9];
  const float* gamma = (const float*)d_in[10];
  const float* beta  = (const float*)d_in[11];
  const float* fc1d  = (const float*)d_in[12];

  float* out    = (float*)d_out;                 // [64][16][256]
  float* logits = out + 64 * 16 * 256;           // [64][16][1000]

  char* ws = (char*)d_ws;
  const size_t OFF_Z     = 0;                    // 31,457,280 B  z f32 [p][b][c][t]
  const size_t OFF_W31A  = 31457280;             //  1,572,864 B  f16 [p][64][768]
  const size_t OFF_W33A  = 33030144;             //  1,572,864 B
  const size_t OFF_W31B  = 34603008;             //    524,288 B  f16 [p][256][64]
  const size_t OFF_W33B  = 35127296;             //  1,572,864 B  f16 [p][256][192]
  const size_t OFF_FEAT  = 36700160;             //    524,288 B  f16 [p][64][256]
  const size_t OFF_FF16T = 37224448;             //  2,097,152 B  f16 [p][256(o)][256(c)]
  const size_t OFF_BNN   = 39321600;             //    524,288 B  f16 [p][64][256]
  const size_t OFF_FCT   = 39845888;             //  8,257,536 B  f16 [p][1008][256]

  float* zg     = (float*)(ws + OFF_Z);
  h16* w31a16   = (h16*)(ws + OFF_W31A);
  h16* w33a16   = (h16*)(ws + OFF_W33A);
  h16* w31b16   = (h16*)(ws + OFF_W31B);
  h16* w33b16   = (h16*)(ws + OFF_W33B);
  h16* feat16   = (h16*)(ws + OFF_FEAT);
  h16* ff16t    = (h16*)(ws + OFF_FF16T);
  h16* bnn16    = (h16*)(ws + OFF_BNN);
  h16* fcT      = (h16*)(ws + OFF_FCT);

  // weight conversions + transposes (tiny); n/4 elements per thread-quad
  cvt_f16_kernel<<<768, 256, 0, stream>>>(w31a, w31a16, 16 * 64 * 768 / 4);
  cvt_f16_kernel<<<768, 256, 0, stream>>>(w33a, w33a16, 16 * 64 * 768 / 4);
  cvt_f16_kernel<<<256, 256, 0, stream>>>(w31b, w31b16, 16 * 256 * 64 / 4);
  cvt_f16_kernel<<<768, 256, 0, stream>>>(w33b, w33b16, 16 * 256 * 192 / 4);
  ff_transpose_kernel<<<4096, 256, 0, stream>>>(FF, ff16t);
  fc_norm_kernel<<<16 * 1008, 256, 0, stream>>>(fc1d, fcT);

  // stage 1: horizontal pool (bandwidth-dominant: 346 MB of x)
  hpool_kernel<<<64 * 256, 256, 0, stream>>>(x, zg);

  // stages 2-4: gated convs (WMMA) + pools + max over T
  gate_feat_kernel<<<16 * 64, 256, 0, stream>>>(zg,
      w31a16, b31a, w31b16, b31b, w33a16, b33a, w33b16, b33b, feat16);

  // stage 5: outputs GEMM (WMMA)
  outputs_gemm_kernel<<<64, 256, 0, stream>>>(feat16, ff16t, out);

  // stage 6: batchnorm + L2 normalize
  bn_norm_kernel<<<16, 256, 0, stream>>>(out, gamma, beta, bnn16);

  // stage 8: logits GEMM (WMMA)
  logits_gemm_kernel<<<512, 256, 0, stream>>>(bnn16, fcT, logits);
}
